// FAVORPlusAttention_13159779795603
// MI455X (gfx1250) — compile-verified, matrix operand order checked by reading, not run on hardware
//
#include <hip/hip_runtime.h>

// ---------------------------------------------------------------------------
// FAVOR+ attention on gfx1250 (MI455X): bf16 WMMA GEMMs + f32 VALU features.
// GEMM: 32x64 tile per wave, K-loop unrolled x2 with ping-pong fragment
// buffers (no register copies; loads of one buffer overlap WMMAs of the
// other). K must be a power of two (it is 1024 here).
// ---------------------------------------------------------------------------

typedef __attribute__((ext_vector_type(8)))  __bf16 v8bf;
typedef __attribute__((ext_vector_type(16))) __bf16 v16bf;
typedef __attribute__((ext_vector_type(8)))  float  v8f;

#define BB 4
#define NN 4096
#define DD 1024
#define HH 16
#define HDIM 64
#define MM 64
#define FEPS 1e-6f

// -------------------- fragment loader (ISA 7.12.2 16-bit A/B layout) -------
// Per lane (l16 = lane&15, hi = lane>>4): element chunks at K offsets
// [hi*8 .. hi*8+7] and [16 + hi*8 .. 16+hi*8+7]  ->  two 16B loads.
static __device__ __forceinline__ v16bf load_frag16(const __bf16* __restrict__ p) {
  v8bf lo = *reinterpret_cast<const v8bf*>(p);
  v8bf hh = *reinterpret_cast<const v8bf*>(p + 16);
  return __builtin_shufflevector(lo, hh, 0, 1, 2, 3, 4, 5, 6, 7,
                                 8, 9, 10, 11, 12, 13, 14, 15);
}

// -------------------- f32 -> bf16 bulk conversion --------------------------
__global__ __launch_bounds__(256) void cvt_f32_to_bf16_kernel(
    const float* __restrict__ in, __bf16* __restrict__ out, int n8) {
  int i = blockIdx.x * blockDim.x + threadIdx.x;
  if (i >= n8) return;
  const float4* p = reinterpret_cast<const float4*>(in) + (size_t)i * 2;
  float4 a = p[0], b = p[1];
  v8bf o;
  o[0] = (__bf16)a.x; o[1] = (__bf16)a.y; o[2] = (__bf16)a.z; o[3] = (__bf16)a.w;
  o[4] = (__bf16)b.x; o[5] = (__bf16)b.y; o[6] = (__bf16)b.z; o[7] = (__bf16)b.w;
  reinterpret_cast<v8bf*>(out)[i] = o;
}

// -------------------- zero init (d_ws is poisoned) -------------------------
__global__ void zero_f32_kernel(float* __restrict__ p, int n) {
  int i = blockIdx.x * blockDim.x + threadIdx.x;
  if (i < n) p[i] = 0.0f;
}

// -------------------- WMMA GEMM: C = A * W^T + bias ------------------------
// A: [rows x K] bf16 row-major, W: [cols x K] bf16 row-major (both K-contig).
// Wave computes a 32x64 tile: 2 A-frags x 4 B-frags -> 8 WMMAs per K-step.
// mode 0: o0[row*cols + col] = val        (final projection -> d_out)
// mode 1: scatter into q/k/v [B,H,N,64]   (QKV GEMM)
__global__ __launch_bounds__(256) void gemm_bf16_wmma_kernel(
    const __bf16* __restrict__ A, const __bf16* __restrict__ W,
    const float* __restrict__ bias, int K, int colTiles, int mode,
    float* __restrict__ o0, float* __restrict__ o1, float* __restrict__ o2) {
  const int lane = threadIdx.x & 31;
  const int wave = blockIdx.x * (blockDim.x >> 5) + (threadIdx.x >> 5);
  const int i0 = (wave / colTiles) << 5;   // 32-row tile
  const int j0 = (wave % colTiles) << 6;   // 64-col strip
  const int l16 = lane & 15;
  const int hi8 = (lane >> 4) << 3;

  const __bf16* __restrict__ pA[2];
  pA[0] = A + (size_t)(i0 + l16) * K + hi8;
  pA[1] = A + (size_t)(i0 + 16 + l16) * K + hi8;
  const __bf16* __restrict__ pB[4];
#pragma unroll
  for (int t = 0; t < 4; ++t)
    pB[t] = W + (size_t)(j0 + t * 16 + l16) * K + hi8;

  v8f acc[2][4] = {};

  // ping-pong fragment buffers (no copies between them)
  v16bf aA[2], bA[4], aB[2], bB[4];
#pragma unroll
  for (int r = 0; r < 2; ++r) aA[r] = load_frag16(pA[r]);
#pragma unroll
  for (int t = 0; t < 4; ++t) bA[t] = load_frag16(pB[t]);

  const int Kmask = K - 1;                 // K is a power of two
  for (int k0 = 0; k0 < K; k0 += 64) {
    const int k1 = k0 + 32;                // always < K (K multiple of 64)
    const int k2 = (k0 + 64) & Kmask;      // wraps to 0 on last iter (harmless)

    // stage k1 into buffer B; overlaps with WMMAs on buffer A
#pragma unroll
    for (int r = 0; r < 2; ++r) aB[r] = load_frag16(pA[r] + k1);
#pragma unroll
    for (int t = 0; t < 4; ++t) bB[t] = load_frag16(pB[t] + k1);
    __builtin_prefetch((const void*)(pA[0] + k0 + 1024), 0, 3);

#pragma unroll
    for (int t = 0; t < 4; ++t) {
      acc[0][t] = __builtin_amdgcn_wmma_f32_16x16x32_bf16(
          false, aA[0], false, bA[t], (short)0, acc[0][t], false, false);
      acc[1][t] = __builtin_amdgcn_wmma_f32_16x16x32_bf16(
          false, aA[1], false, bA[t], (short)0, acc[1][t], false, false);
    }

    // stage k2 into buffer A; overlaps with WMMAs on buffer B
#pragma unroll
    for (int r = 0; r < 2; ++r) aA[r] = load_frag16(pA[r] + k2);
#pragma unroll
    for (int t = 0; t < 4; ++t) bA[t] = load_frag16(pB[t] + k2);

#pragma unroll
    for (int t = 0; t < 4; ++t) {
      acc[0][t] = __builtin_amdgcn_wmma_f32_16x16x32_bf16(
          false, aB[0], false, bB[t], (short)0, acc[0][t], false, false);
      acc[1][t] = __builtin_amdgcn_wmma_f32_16x16x32_bf16(
          false, aB[1], false, bB[t], (short)0, acc[1][t], false, false);
    }
  }

  // C layout: VGPR r holds rows M=r (lanes 0-15) and M=r+8 (lanes 16-31).
#pragma unroll
  for (int ri = 0; ri < 2; ++ri) {
    const int rbase = i0 + ri * 16 + hi8;
#pragma unroll
    for (int t = 0; t < 4; ++t) {
      const int col = j0 + t * 16 + l16;
      const float bia = bias[col];
#pragma unroll
      for (int r = 0; r < 8; ++r) {
        const int row = rbase + r;
        const float val = acc[ri][t][r] + bia;
        if (mode == 0) {
          o0[(size_t)row * (size_t)(colTiles << 6) + col] = val;
        } else {
          const int s  = col >> 10;          // 0=q, 1=k, 2=v
          const int h  = (col >> 6) & (HH - 1);
          const int hd = col & (HDIM - 1);
          const int b  = row >> 12;          // row / N
          const int n  = row & (NN - 1);
          float* __restrict__ dst = (s == 0) ? o0 : ((s == 1) ? o1 : o2);
          dst[((size_t)(b * HH + h) * NN + n) * HDIM + hd] = val;
        }
      }
    }
  }
}

// -------------------- FAVOR+ feature map -----------------------------------
// dst[row,m] = exp(comb - rowmax) / sqrt(M) + eps,
// comb = (src/8) . pm[m] - 0.5*||src/8||^2.  One thread per row (n).
__global__ __launch_bounds__(256) void favor_features_kernel(
    const float* __restrict__ src, const float* __restrict__ pm,
    float* __restrict__ dst) {
  __shared__ float lds_pm[MM * HDIM];
  for (int t = threadIdx.x; t < MM * HDIM; t += 256) lds_pm[t] = pm[t];
  __syncthreads();

  const int gid = blockIdx.x * 256 + threadIdx.x;      // row in [0, B*H*N)
  const float* __restrict__ row = src + (size_t)gid * HDIM;

  float dn[HDIM];
#pragma unroll
  for (int d = 0; d < HDIM; ++d) dn[d] = row[d] * 0.125f;   // 1/sqrt(64)
  float nsq = 0.0f;
#pragma unroll
  for (int d = 0; d < HDIM; ++d) nsq += dn[d] * dn[d];
  nsq *= 0.5f;

  float mx = -1e30f;
  for (int m = 0; m < MM; ++m) {
    float c = -nsq;
#pragma unroll
    for (int d = 0; d < HDIM; ++d) c += dn[d] * lds_pm[m * HDIM + d];
    mx = fmaxf(mx, c);
  }
  float* __restrict__ out = dst + (size_t)gid * MM;
  for (int m = 0; m < MM; ++m) {
    float c = -nsq;
#pragma unroll
    for (int d = 0; d < HDIM; ++d) c += dn[d] * lds_pm[m * HDIM + d];
    out[m] = __expf(c - mx) * 0.125f + FEPS;           // 1/sqrt(M)=0.125
  }
}

// -------------------- kv[m,v] = sum_n k_p[n,m]*v[n,v]; k_sum[m] ------------
// Split-K over N (8 partitions) with f32 global atomics. LDS-tiled.
__global__ __launch_bounds__(256) void kv_ksum_kernel(
    const float* __restrict__ kp, const float* __restrict__ vsrc,
    float* __restrict__ kv, float* __restrict__ ksum) {
  const int bh = blockIdx.x >> 3;
  const int part = blockIdx.x & 7;
  const int tid = threadIdx.x;
  const int m0 = tid >> 2;
  const int v0 = (tid & 3) << 4;
  __shared__ float lk[32 * MM];
  __shared__ float lv[32 * HDIM];

  float acc[16] = {};
  float ks = 0.0f;
  const long base = (long)bh * NN * HDIM + (long)part * 512 * HDIM;

  for (int n0 = 0; n0 < 512; n0 += 32) {
    __syncthreads();
    for (int t = tid; t < 32 * 64; t += 256) {
      lk[t] = kp[base + (long)n0 * MM + t];
      lv[t] = vsrc[base + (long)n0 * HDIM + t];
    }
    __syncthreads();
    for (int n = 0; n < 32; ++n) {
      const float kk = lk[n * MM + m0];
      ks += kk;
#pragma unroll
      for (int vv = 0; vv < 16; ++vv) acc[vv] += kk * lv[n * HDIM + v0 + vv];
    }
  }
#pragma unroll
  for (int vv = 0; vv < 16; ++vv)
    atomicAdd(&kv[(long)bh * MM * HDIM + m0 * HDIM + v0 + vv], acc[vv]);
  if ((tid & 3) == 0) atomicAdd(&ksum[bh * MM + m0], ks);
}

// -------------------- out[n,v] = (q_p[n,:] @ kv[:,v]) / (q_p . k_sum + eps) -
// Writes the attention output directly as bf16 in [B,N,D] for the proj GEMM.
__global__ __launch_bounds__(256) void favor_out_kernel(
    const float* __restrict__ qp, const float* __restrict__ kv,
    const float* __restrict__ ksum, __bf16* __restrict__ attn_bf) {
  const int bh = blockIdx.x >> 6;          // 64 blocks of 64 rows per (b,h)
  const int nblk = blockIdx.x & 63;
  const int tid = threadIdx.x;
  const int nl = tid >> 2;
  const int v0 = (tid & 3) << 4;
  __shared__ float lkv[MM * HDIM];
  __shared__ float lks[MM];
  for (int t = tid; t < MM * HDIM; t += 256) lkv[t] = kv[(long)bh * MM * HDIM + t];
  if (tid < MM) lks[tid] = ksum[bh * MM + tid];
  __syncthreads();

  const int n = nblk * 64 + nl;
  const float* __restrict__ qrow = qp + ((long)bh * NN + n) * MM;
  float o[16] = {};
  float norm = 0.0f;
  for (int m = 0; m < MM; ++m) {
    const float qm = qrow[m];
    norm += qm * lks[m];
#pragma unroll
    for (int vv = 0; vv < 16; ++vv) o[vv] += qm * lkv[m * HDIM + v0 + vv];
  }
  const float inv = 1.0f / (norm + FEPS);
  const int b = bh >> 4, h = bh & (HH - 1);
  __bf16* __restrict__ dst = attn_bf + ((long)b * NN + n) * DD + h * HDIM + v0;
#pragma unroll
  for (int vv = 0; vv < 16; ++vv) dst[vv] = (__bf16)(o[vv] * inv);
}

// ===========================================================================
extern "C" void kernel_launch(void* const* d_in, const int* in_sizes, int n_in,
                              void* d_out, int out_size, void* d_ws, size_t ws_size,
                              hipStream_t stream) {
  const float* x      = (const float*)d_in[0];
  const float* qkv_w  = (const float*)d_in[1];
  const float* qkv_b  = (const float*)d_in[2];
  const float* proj_w = (const float*)d_in[3];
  const float* proj_b = (const float*)d_in[4];
  const float* pm     = (const float*)d_in[5];
  float* out = (float*)d_out;

  // ---- workspace layout -------------------------------------------------
  char* ws = (char*)d_ws;
  size_t off = 0;
  auto walloc = [&](size_t bytes) -> void* {
    void* p = ws + off;
    off += (bytes + 255) & ~(size_t)255;
    return p;
  };
  const size_t nX  = (size_t)BB * NN * DD;        // 16.7M
  const size_t nHD = (size_t)BB * HH * NN * HDIM; // 16.7M (also = B*H*N*M)

  __bf16* x_bf     = (__bf16*)walloc(nX * 2);
  __bf16* wqkv_bf  = (__bf16*)walloc((size_t)3 * DD * DD * 2);
  __bf16* wproj_bf = (__bf16*)walloc((size_t)DD * DD * 2);
  float*  qf       = (float*)walloc(nHD * 4);
  float*  kf       = (float*)walloc(nHD * 4);
  float*  vf       = (float*)walloc(nHD * 4);
  float*  q_p      = (float*)walloc(nHD * 4);
  float*  k_p      = (float*)walloc(nHD * 4);
  float*  kv       = (float*)walloc((size_t)BB * HH * MM * HDIM * 4);
  float*  ksum     = (float*)walloc((size_t)BB * HH * MM * 4);   // contiguous after kv
  __bf16* attn_bf  = (__bf16*)walloc(nX * 2);
  (void)ws_size; (void)in_sizes; (void)n_in; (void)out_size;

  // ---- 1) bf16 copies of x and weights ---------------------------------
  cvt_f32_to_bf16_kernel<<<(int)(nX / 8 / 256), 256, 0, stream>>>(x, x_bf, (int)(nX / 8));
  cvt_f32_to_bf16_kernel<<<3 * DD * DD / 8 / 256, 256, 0, stream>>>(qkv_w, wqkv_bf, 3 * DD * DD / 8);
  cvt_f32_to_bf16_kernel<<<DD * DD / 8 / 256, 256, 0, stream>>>(proj_w, wproj_bf, DD * DD / 8);

  // ---- 2) QKV GEMM (WMMA): waves = (16384/32)*(3072/64) = 24576 ---------
  gemm_bf16_wmma_kernel<<<24576 / 8, 256, 0, stream>>>(
      x_bf, wqkv_bf, qkv_b, DD, 3 * DD / 64, /*mode=*/1, qf, kf, vf);

  // ---- 3) FAVOR features for q and k ------------------------------------
  favor_features_kernel<<<(int)(nHD / HDIM / 256), 256, 0, stream>>>(qf, pm, q_p);
  favor_features_kernel<<<(int)(nHD / HDIM / 256), 256, 0, stream>>>(kf, pm, k_p);

  // ---- 4) kv + k_sum (zero first: ws is poisoned) -----------------------
  const int nZero = BB * HH * MM * HDIM + BB * HH * MM;   // kv + ksum contiguous
  zero_f32_kernel<<<(nZero + 255) / 256, 256, 0, stream>>>(kv, nZero);
  kv_ksum_kernel<<<BB * HH * 8, 256, 0, stream>>>(k_p, vf, kv, ksum);

  // ---- 5) normalized attention output (bf16) ----------------------------
  favor_out_kernel<<<BB * HH * (NN / 64), 256, 0, stream>>>(q_p, kv, ksum, attn_bf);

  // ---- 6) output projection GEMM (WMMA) -> d_out ------------------------
  gemm_bf16_wmma_kernel<<<8192 / 8, 256, 0, stream>>>(
      attn_bf, wproj_bf, proj_b, DD, DD / 64, /*mode=*/0, out, nullptr, nullptr);
}
